// Model_31903017075250
// MI455X (gfx1250) — matrix operand order repaired and checked by reading, not compiled, not run on hardware
//
#include <hip/hip_runtime.h>
#include <cstdint>
#include <cstddef>

typedef __bf16 bf16;
typedef __attribute__((ext_vector_type(16))) __bf16 v16bf;
typedef __attribute__((ext_vector_type(8)))  float  v8f;

struct U32x8 { uint4 lo, hi; };

#define B_SZ    32
#define SEQ     720
#define ENC_IN  862
#define D_MODEL 256
#define PRED    720
#define D_STATE 16
#define D_CONV  4
#define D_INNER 256
#define DT_RANK 16
#define NROWS   (B_SZ * ENC_IN)   // 27584 tokens, multiple of 16
#define KPAD    736               // 720 padded to multiple of 32

// ---------------------------------------------------------------------------
// Weight convert f32 -> bf16 (row-major N x K), with optional K zero-padding.
// ---------------------------------------------------------------------------
__global__ void cvt_pad_kernel(const float* __restrict__ src, bf16* __restrict__ dst,
                               int rows, int cols, int ldd) {
    int i = blockIdx.x * blockDim.x + threadIdx.x;
    int total = rows * ldd;
    if (i >= total) return;
    int r = i / ldd, c = i - r * ldd;
    dst[i] = (c < cols) ? (bf16)src[r * cols + c] : (bf16)0.0f;
}

// ---------------------------------------------------------------------------
// RevIN stats + normalize + (B,SEQ,ENC_IN) -> (B*ENC_IN, KPAD) bf16 transpose.
// ---------------------------------------------------------------------------
__global__ __launch_bounds__(64)
void revin_pack_kernel(const float* __restrict__ x,
                       const float* __restrict__ rw, const float* __restrict__ rb,
                       bf16* __restrict__ xn,
                       float* __restrict__ meanv, float* __restrict__ stdv) {
    int b = blockIdx.y;
    int c = blockIdx.x * 64 + threadIdx.x;
    if (c >= ENC_IN) return;
    const float* p = x + (size_t)b * SEQ * ENC_IN + c;
    float s = 0.f, ss = 0.f;
    for (int t = 0; t < SEQ; ++t) {
        float v = p[(size_t)t * ENC_IN];
        s += v; ss += v * v;
    }
    float mean = s * (1.0f / SEQ);
    float var  = ss * (1.0f / SEQ) - mean * mean;
    float sd   = sqrtf(var + 1e-5f);
    int row = b * ENC_IN + c;
    meanv[row] = mean;
    stdv[row]  = sd;
    float w = rw[c], bb = rb[c], inv = 1.0f / sd;
    bf16* q = xn + (size_t)row * KPAD;
    for (int t = 0; t < SEQ; ++t)
        q[t] = (bf16)(((p[(size_t)t * ENC_IN] - mean) * inv) * w + bb);
    for (int t = SEQ; t < KPAD; ++t) q[t] = (bf16)0.0f;
}

// ---------------------------------------------------------------------------
// WMMA bf16 GEMM: C(MxN) = A(MxK, row major bf16) * W^T (W is N x K row major,
// row stride == K, K compile-time). Rolled K-loop with PING-PONG register
// double-buffering, two slabs per iteration: slab kk+1 loads into buffer B at
// immediate offset +32 while buffer A's WMMAs issue; pointers advance once
// (+64) per iteration; slab kk+2 loads into buffer A while B's WMMAs issue.
// No buffer-rotation moves. Tail parity resolved at compile time.
// Block = 128 threads (4 waves); wave = one 16-row tile x 64 cols (4 accs).
// ---------------------------------------------------------------------------
struct GemmEpi {
    int          mode;     // 0: bf16 out (+bias)  1: f32 out  3: lin4 + de-RevIN
    bf16*        out_bf;
    float*       out_f;
    int          ldo;
    const float* bias;
    // mode 3 extras
    float*       dout;
    const float* rwp;
    const float* rbp;
    const float* stdp;
    const float* meanp;
};

__device__ __forceinline__ v16bf load_afrag(const bf16* arow) {
    const uint4* pa0 = (const uint4*)(arow);
    const uint4* pa1 = (const uint4*)(arow + 8);
    U32x8 t; t.lo = pa0[0]; t.hi = pa1[0];
    return __builtin_bit_cast(v16bf, t);
}
__device__ __forceinline__ v16bf load_bfrag(const bf16* wrow) {
    const uint4* pb = (const uint4*)(wrow);
    U32x8 t; t.lo = pb[0]; t.hi = pb[1];
    return __builtin_bit_cast(v16bf, t);
}
#define WMMA_BF16(a, b, c) __builtin_amdgcn_wmma_f32_16x16x32_bf16( \
        false, (a), false, (b), (short)0, (c), false, false)

template <int K>
__global__ __launch_bounds__(128)
void gemm_wmma_kernel(const bf16* __restrict__ A,
                      const bf16* __restrict__ W,
                      int M, int N, GemmEpi e) {
    const int wave  = threadIdx.x >> 5;
    const int lane  = threadIdx.x & 31;
    const int mtile = blockIdx.y * 4 + wave;
    const int row0  = mtile * 16;
    const int ncol0 = blockIdx.x * 64;
    if (row0 >= M) return;

    const int m      = lane & 15;
    const int hi     = (lane >> 4) & 1;
    const int kaBase = hi * 16;   // A fragment: 16 consecutive K values
    const int kbBase = hi * 16;   // B fragment: 16 consecutive K values

    // Base pointers (B rows clamped; stores guarded later).
    const bf16* pa = A + (size_t)(row0 + m) * K + kaBase;
    int n0 = ncol0 + 0 * 16 + (lane & 15); if (n0 >= N) n0 = N - 1;
    int n1 = ncol0 + 1 * 16 + (lane & 15); if (n1 >= N) n1 = N - 1;
    int n2 = ncol0 + 2 * 16 + (lane & 15); if (n2 >= N) n2 = N - 1;
    int n3 = ncol0 + 3 * 16 + (lane & 15); if (n3 >= N) n3 = N - 1;
    const bf16* pw0 = W + (size_t)n0 * K + kbBase;
    const bf16* pw1 = W + (size_t)n1 * K + kbBase;
    const bf16* pw2 = W + (size_t)n2 * K + kbBase;
    const bf16* pw3 = W + (size_t)n3 * K + kbBase;

    v8f acc0, acc1, acc2, acc3;
#pragma unroll
    for (int r = 0; r < 8; ++r) { acc0[r] = 0.f; acc1[r] = 0.f; acc2[r] = 0.f; acc3[r] = 0.f; }

    constexpr int nk      = K >> 5;        // #32-wide K slabs (23 or 8)
    constexpr int nsteady = (nk - 1) / 2;  // 2-slab iterations
    constexpr bool oddTail = (((nk - 1) & 1) == 1);

    // Prologue: slab 0 -> buffer A.
    v16bf aA  = load_afrag(pa);
    v16bf bA0 = load_bfrag(pw0);
    v16bf bA1 = load_bfrag(pw1);
    v16bf bA2 = load_bfrag(pw2);
    v16bf bA3 = load_bfrag(pw3);
    v16bf aB, bB0, bB1, bB2, bB3;

#pragma unroll 1
    for (int it = 0; it < nsteady; ++it) {
        // Half 1: load slab kk+1 -> B (immediate +32 elems), consume A.
        __builtin_prefetch(pa + 160, 0, 1);
        aB  = load_afrag(pa  + 32);
        bB0 = load_bfrag(pw0 + 32);
        bB1 = load_bfrag(pw1 + 32);
        bB2 = load_bfrag(pw2 + 32);
        bB3 = load_bfrag(pw3 + 32);
        acc0 = WMMA_BF16(aA, bA0, acc0);
        acc1 = WMMA_BF16(aA, bA1, acc1);
        acc2 = WMMA_BF16(aA, bA2, acc2);
        acc3 = WMMA_BF16(aA, bA3, acc3);
        // Half 2: advance to slab kk+2, load -> A, consume B.
        pa += 64; pw0 += 64; pw1 += 64; pw2 += 64; pw3 += 64;
        aA  = load_afrag(pa);
        bA0 = load_bfrag(pw0);
        bA1 = load_bfrag(pw1);
        bA2 = load_bfrag(pw2);
        bA3 = load_bfrag(pw3);
        acc0 = WMMA_BF16(aB, bB0, acc0);
        acc1 = WMMA_BF16(aB, bB1, acc1);
        acc2 = WMMA_BF16(aB, bB2, acc2);
        acc3 = WMMA_BF16(aB, bB3, acc3);
    }
    if constexpr (oddTail) {
        // One more slab to load (buffer A holds slab nk-2).
        aB  = load_afrag(pa  + 32);
        bB0 = load_bfrag(pw0 + 32);
        bB1 = load_bfrag(pw1 + 32);
        bB2 = load_bfrag(pw2 + 32);
        bB3 = load_bfrag(pw3 + 32);
        acc0 = WMMA_BF16(aA, bA0, acc0);
        acc1 = WMMA_BF16(aA, bA1, acc1);
        acc2 = WMMA_BF16(aA, bA2, acc2);
        acc3 = WMMA_BF16(aA, bA3, acc3);
        acc0 = WMMA_BF16(aB, bB0, acc0);
        acc1 = WMMA_BF16(aB, bB1, acc1);
        acc2 = WMMA_BF16(aB, bB2, acc2);
        acc3 = WMMA_BF16(aB, bB3, acc3);
    } else {
        // Buffer A holds the final slab.
        acc0 = WMMA_BF16(aA, bA0, acc0);
        acc1 = WMMA_BF16(aA, bA1, acc1);
        acc2 = WMMA_BF16(aA, bA2, acc2);
        acc3 = WMMA_BF16(aA, bA3, acc3);
    }

    // Store. C layout: VGPR r -> M = r + hi*8, N = lane&15.
    const int mOff = hi * 8;
    v8f accs[4] = {acc0, acc1, acc2, acc3};
#pragma unroll
    for (int j = 0; j < 4; ++j) {
        int n = ncol0 + j * 16 + (lane & 15);
        if (n >= N) continue;
        float bv = e.bias ? e.bias[n] : 0.0f;
#pragma unroll
        for (int r = 0; r < 8; ++r) {
            int    mm = row0 + mOff + r;
            float  v  = accs[j][r] + bv;
            if (e.mode == 0) {
                e.out_bf[(size_t)mm * e.ldo + n] = (bf16)v;
            } else if (e.mode == 1) {
                e.out_f[(size_t)mm * e.ldo + n] = v;
            } else { // mode 3: de-RevIN + transposed store to (B, PRED, ENC_IN)
                int   b  = mm / ENC_IN;
                int   c  = mm - b * ENC_IN;
                float ov = (v - e.rbp[c]) / (e.rwp[c] + 1e-10f) * e.stdp[mm] + e.meanp[mm];
                e.dout[((size_t)b * PRED + n) * ENC_IN + c] = ov;
            }
        }
    }
}

// ---------------------------------------------------------------------------
// Depthwise causal conv (k=4) along the 862-token axis + bias + SiLU.
// xz layout: (NROWS, 512) f32, u part = cols [0,256). Output u: (NROWS, 256).
// ---------------------------------------------------------------------------
__global__ void dwconv_silu_kernel(const float* __restrict__ xz,
                                   const float* __restrict__ cw,
                                   const float* __restrict__ cb,
                                   float* __restrict__ u) {
    size_t i = (size_t)blockIdx.x * blockDim.x + threadIdx.x;
    if (i >= (size_t)NROWS * D_INNER) return;
    int    d   = (int)(i & (D_INNER - 1));
    size_t row = i >> 8;
    int    l   = (int)(row % ENC_IN);
    float acc = cb[d];
#pragma unroll
    for (int j = 0; j < D_CONV; ++j) {
        int ls = l - (D_CONV - 1) + j;
        if (ls >= 0)
            acc += cw[d * D_CONV + j] * xz[(row - (size_t)(D_CONV - 1 - j)) * 512 + d];
    }
    float s = acc / (1.0f + __expf(-acc));     // SiLU
    u[row * D_INNER + d] = s;
}

// ---------------------------------------------------------------------------
// x_proj (256->48) + dt projection (16->256) + softplus. One block per token.
// ---------------------------------------------------------------------------
__global__ __launch_bounds__(64)
void xproj_dt_kernel(const float* __restrict__ u,
                     const float* __restrict__ xw,   // (48, 256)
                     const float* __restrict__ dtw,  // (256, 16)
                     const float* __restrict__ dtb,
                     float* __restrict__ delta,
                     float* __restrict__ Bm, float* __restrict__ Cm) {
    __shared__ float su[D_INNER];
    __shared__ float dbl[DT_RANK + 2 * D_STATE];
    size_t row = blockIdx.x;
    int    tid = threadIdx.x;
    for (int i = tid; i < D_INNER; i += 64) su[i] = u[row * D_INNER + i];
    __syncthreads();
    if (tid < DT_RANK + 2 * D_STATE) {
        float a = 0.f;
        const float* wr = xw + tid * D_INNER;
        for (int k = 0; k < D_INNER; ++k) a += wr[k] * su[k];
        dbl[tid] = a;
    }
    __syncthreads();
    if (tid < D_STATE) {
        Bm[row * D_STATE + tid] = dbl[DT_RANK + tid];
        Cm[row * D_STATE + tid] = dbl[DT_RANK + D_STATE + tid];
    }
    for (int d = tid; d < D_INNER; d += 64) {
        float a = dtb[d];
        const float* wr = dtw + d * DT_RANK;
#pragma unroll
        for (int r = 0; r < DT_RANK; ++r) a += wr[r] * dbl[r];
        delta[row * D_INNER + d] = (a > 20.0f) ? a : log1pf(__expf(a));
    }
}

// ---------------------------------------------------------------------------
// Selective scan. thread = (d_local*16 + n); 16-lane shfl_xor reduction for
// y = sum_n h[n]*C[n]. Fuses +u*D and the silu(z) gate; outputs bf16.
// ---------------------------------------------------------------------------
__global__ __launch_bounds__(1024)
void scan_kernel(const float* __restrict__ delta, const float* __restrict__ Bm,
                 const float* __restrict__ Cm,    const float* __restrict__ u,
                 const float* __restrict__ xz,    const float* __restrict__ A_log,
                 const float* __restrict__ Dv,    bf16* __restrict__ ys) {
    int b = blockIdx.x;                       // batch
    int n = threadIdx.x & 15;                 // state index
    int d = blockIdx.y * 64 + (threadIdx.x >> 4);
    float a  = -__expf(A_log[d * D_STATE + n]);
    float Dd = Dv[d];
    float h  = 0.0f;
    size_t rowBase = (size_t)b * ENC_IN;
    for (int t = 0; t < ENC_IN; ++t) {
        size_t row = rowBase + t;
        float dl = delta[row * D_INNER + d];
        float ut = u[row * D_INNER + d];
        float bn = Bm[row * D_STATE + n];
        float cn = Cm[row * D_STATE + n];
        float dA = __expf(dl * a);
        h = dA * h + (dl * bn) * ut;
        float p = h * cn;
        p += __shfl_xor(p, 1);
        p += __shfl_xor(p, 2);
        p += __shfl_xor(p, 4);
        p += __shfl_xor(p, 8);
        if (n == 0) {
            float z  = xz[row * 512 + D_INNER + d];
            float sz = z / (1.0f + __expf(-z));
            ys[row * D_INNER + d] = (bf16)((p + ut * Dd) * sz);
        }
    }
}

// ---------------------------------------------------------------------------
extern "C" void kernel_launch(void* const* d_in, const int* in_sizes, int n_in,
                              void* d_out, int out_size, void* d_ws, size_t ws_size,
                              hipStream_t stream) {
    const float* x_enc   = (const float*)d_in[0];
    const float* revin_w = (const float*)d_in[4];
    const float* revin_b = (const float*)d_in[5];
    const float* lin1_w  = (const float*)d_in[6];
    const float* lin1_b  = (const float*)d_in[7];
    const float* lin4_w  = (const float*)d_in[8];
    const float* lin4_b  = (const float*)d_in[9];
    const float* in_proj = (const float*)d_in[10];
    const float* conv_w  = (const float*)d_in[11];
    const float* conv_b  = (const float*)d_in[12];
    const float* x_projw = (const float*)d_in[13];
    const float* dt_w    = (const float*)d_in[14];
    const float* dt_b    = (const float*)d_in[15];
    const float* A_log   = (const float*)d_in[16];
    const float* Dvec    = (const float*)d_in[17];
    const float* out_prj = (const float*)d_in[18];
    float* out = (float*)d_out;

    // bump allocator on workspace
    uintptr_t p = (uintptr_t)d_ws;
    auto alloc = [&](size_t bytes) -> void* {
        uintptr_t r = (p + 255) & ~(uintptr_t)255;
        p = r + bytes;
        return (void*)r;
    };
    float* meanv = (float*)alloc(sizeof(float) * NROWS);
    float* stdv  = (float*)alloc(sizeof(float) * NROWS);
    bf16*  xn    = (bf16*) alloc(sizeof(bf16) * (size_t)NROWS * KPAD);
    bf16*  w1    = (bf16*) alloc(sizeof(bf16) * (size_t)D_MODEL * KPAD);
    bf16*  wip   = (bf16*) alloc(sizeof(bf16) * (size_t)(2 * D_INNER) * D_MODEL);
    bf16*  wop   = (bf16*) alloc(sizeof(bf16) * (size_t)D_MODEL * D_INNER);
    bf16*  w4    = (bf16*) alloc(sizeof(bf16) * (size_t)PRED * D_MODEL);
    bf16*  x0    = (bf16*) alloc(sizeof(bf16) * (size_t)NROWS * D_MODEL);
    float* xz    = (float*)alloc(sizeof(float) * (size_t)NROWS * 2 * D_INNER);
    float* uu    = (float*)alloc(sizeof(float) * (size_t)NROWS * D_INNER);
    float* delta = (float*)alloc(sizeof(float) * (size_t)NROWS * D_INNER);
    float* Bm    = (float*)alloc(sizeof(float) * (size_t)NROWS * D_STATE);
    float* Cm    = (float*)alloc(sizeof(float) * (size_t)NROWS * D_STATE);
    bf16*  ysb   = (bf16*) alloc(sizeof(bf16) * (size_t)NROWS * D_INNER);
    bf16*  mo    = (bf16*) alloc(sizeof(bf16) * (size_t)NROWS * D_INNER);
    (void)ws_size; (void)in_sizes; (void)n_in; (void)out_size;

    // 1) weight conversion to bf16 (K-padded where needed)
    auto cvt = [&](const float* s, bf16* d, int rows, int cols, int ldd) {
        int tot = rows * ldd;
        cvt_pad_kernel<<<(tot + 255) / 256, 256, 0, stream>>>(s, d, rows, cols, ldd);
    };
    cvt(lin1_w,  w1,  D_MODEL,     SEQ,     KPAD);
    cvt(in_proj, wip, 2 * D_INNER, D_MODEL, D_MODEL);
    cvt(out_prj, wop, D_MODEL,     D_INNER, D_INNER);
    cvt(lin4_w,  w4,  PRED,        D_MODEL, D_MODEL);

    // 2) RevIN + transpose + bf16 pack
    revin_pack_kernel<<<dim3((ENC_IN + 63) / 64, B_SZ), 64, 0, stream>>>(
        x_enc, revin_w, revin_b, xn, meanv, stdv);

    dim3 blk(128);
    // 3) lin1: (27584 x 736) x (736 x 256) -> x0 bf16
    {
        GemmEpi e{}; e.mode = 0; e.out_bf = x0; e.ldo = D_MODEL; e.bias = lin1_b;
        gemm_wmma_kernel<KPAD><<<dim3((D_MODEL + 63) / 64, NROWS / 64), blk, 0, stream>>>(
            xn, w1, NROWS, D_MODEL, e);
    }
    // 4) in_proj: (27584 x 256) x (256 x 512) -> xz f32
    {
        GemmEpi e{}; e.mode = 1; e.out_f = xz; e.ldo = 2 * D_INNER; e.bias = nullptr;
        gemm_wmma_kernel<D_MODEL><<<dim3((2 * D_INNER + 63) / 64, NROWS / 64), blk, 0, stream>>>(
            x0, wip, NROWS, 2 * D_INNER, e);
    }
    // 5) depthwise conv + SiLU
    {
        size_t tot = (size_t)NROWS * D_INNER;
        dwconv_silu_kernel<<<(unsigned)((tot + 255) / 256), 256, 0, stream>>>(
            xz, conv_w, conv_b, uu);
    }
    // 6) x_proj + dt + softplus
    xproj_dt_kernel<<<NROWS, 64, 0, stream>>>(uu, x_projw, dt_w, dt_b, delta, Bm, Cm);

    // 7) selective scan (+ skip + gate), bf16 out
    scan_kernel<<<dim3(B_SZ, D_INNER / 64), 1024, 0, stream>>>(
        delta, Bm, Cm, uu, xz, A_log, Dvec, ysb);

    // 8) out_proj: (27584 x 256) x (256 x 256) -> mo bf16
    {
        GemmEpi e{}; e.mode = 0; e.out_bf = mo; e.ldo = D_MODEL; e.bias = nullptr;
        gemm_wmma_kernel<D_MODEL><<<dim3((D_MODEL + 63) / 64, NROWS / 64), blk, 0, stream>>>(
            ysb, wop, NROWS, D_MODEL, e);
    }
    // 9) lin4 + de-RevIN + transposed store to d_out (B, PRED, ENC_IN)
    {
        GemmEpi e{}; e.mode = 3; e.bias = lin4_b;
        e.dout = out; e.rwp = revin_w; e.rbp = revin_b; e.stdp = stdv; e.meanp = meanv;
        gemm_wmma_kernel<D_MODEL><<<dim3((PRED + 63) / 64, NROWS / 64), blk, 0, stream>>>(
            mo, w4, NROWS, PRED, e);
    }
}
